// VectorizedHypertoroidalAttention_31817117729231
// MI455X (gfx1250) — matrix operand order repaired and checked by reading, not compiled, not run on hardware
//
#include <hip/hip_runtime.h>
#include <hip/hip_bf16.h>

// ---------------------------------------------------------------------------
// VectorizedHypertoroidalAttention for MI455X (gfx1250, wave32, WMMA)
// Pipeline: cvt -> qkv GEMM (bf16 WMMA) -> LN+splat (VALU) -> flash attn
// (bf16 WMMA, online softmax, K=32 padded rank-20 logits) -> out GEMM.
// Workspace use: ~82 MiB.
// ---------------------------------------------------------------------------

typedef __bf16 bf16_t;
typedef __attribute__((ext_vector_type(16))) __bf16 v16bf;
typedef __attribute__((ext_vector_type(8)))  float  v8f;

union FragU { v16bf v; uint4 q[2]; };

// A-matrix 16x32 bf16 fragment (ISA 7.12.2): lane L holds row m=L&15,
// K-halves: V0..3 -> K = 8*hi + {0..7}, V4..7 -> K = 16 + 8*hi + {0..7}.
__device__ __forceinline__ v16bf load_fragA(const bf16_t* base, int r0, int ld,
                                            int k0, int lane) {
  int row = r0 + (lane & 15);
  int hi  = lane >> 4;
  const bf16_t* p = base + (size_t)row * ld + k0 + hi * 8;
  FragU f;
  f.q[0] = *(const uint4*)(p);
  f.q[1] = *(const uint4*)(p + 16);
  return f.v;
}

// B-matrix 32x16 bf16 fragment: lane L holds column n=L&15;
// V0..3 -> K = 8*hi + {0..7}, V4..7 -> K = 16 + 8*hi + {0..7} (matches A).
__device__ __forceinline__ v16bf load_fragB(const bf16_t* base, int c0, int ld,
                                            int k0, int lane) {
  int col = c0 + (lane & 15);
  int hi  = lane >> 4;
  const bf16_t* p = base + (size_t)col * ld + k0 + hi * 8;
  FragU f;
  f.q[0] = *(const uint4*)(p);
  f.q[1] = *(const uint4*)(p + 16);
  return f.v;
}

__device__ __forceinline__ v8f wmma_bf16(v16bf a, v16bf b, v8f c) {
  return __builtin_amdgcn_wmma_f32_16x16x32_bf16(false, a, false, b,
                                                 (short)0, c, false, false);
}

// ---------------------------------------------------------------------------
// K1: f32 -> bf16 convert
// ---------------------------------------------------------------------------
__global__ void cvt_bf16_kernel(const float* __restrict__ in,
                                bf16_t* __restrict__ out, int n) {
  int i = blockIdx.x * blockDim.x + threadIdx.x;
  if (i < n) out[i] = (bf16_t)in[i];
}

// ---------------------------------------------------------------------------
// K2/K5: C[M,N] = A[M,K](bf16) @ W[N,K]^T(bf16) + bias, fp32 out.
// Block 256 = 8 waves (2x4), block tile 64x128, wave tile 32x32.
// ---------------------------------------------------------------------------
__global__ __launch_bounds__(256)
void gemm_bf16_kernel(const bf16_t* __restrict__ A, const bf16_t* __restrict__ W,
                      const float* __restrict__ bias, float* __restrict__ C,
                      int M, int N, int K) {
  int lane = threadIdx.x & 31;
  int wid  = threadIdx.x >> 5;
  int wm = wid & 1, wn = wid >> 1;
  int m0 = blockIdx.y * 64 + wm * 32;
  int n0 = blockIdx.x * 128 + wn * 32;

  v8f c00 = {0,0,0,0,0,0,0,0}, c01 = {0,0,0,0,0,0,0,0};
  v8f c10 = {0,0,0,0,0,0,0,0}, c11 = {0,0,0,0,0,0,0,0};

  for (int k0 = 0; k0 < K; k0 += 32) {
    v16bf a0 = load_fragA(A, m0,      K, k0, lane);
    v16bf a1 = load_fragA(A, m0 + 16, K, k0, lane);
    v16bf b0 = load_fragB(W, n0,      K, k0, lane);
    v16bf b1 = load_fragB(W, n0 + 16, K, k0, lane);
    c00 = wmma_bf16(a0, b0, c00);
    c01 = wmma_bf16(a0, b1, c01);
    c10 = wmma_bf16(a1, b0, c10);
    c11 = wmma_bf16(a1, b1, c11);
  }

  int n  = n0 + (lane & 15);
  int mo = m0 + ((lane >> 4) << 3);     // C layout: row = r + 8*hi, col = lane&15
  float bv0 = bias[n], bv1 = bias[n + 16];
#pragma unroll
  for (int r = 0; r < 8; ++r) {
    C[(size_t)(mo + r) * N + n]           = c00[r] + bv0;
    C[(size_t)(mo + r) * N + n + 16]      = c01[r] + bv1;
    C[(size_t)(mo + 16 + r) * N + n]      = c10[r] + bv0;
    C[(size_t)(mo + 16 + r) * N + n + 16] = c11[r] + bv1;
  }
}

// ---------------------------------------------------------------------------
// K3: per-(b,h,t): LN(q), LN(k), torus splat factors, weights folded into Fq.
// Writes Fqw[bh][t][32], Fk[bh][t][32] (c=20..31 zero), Vt[bh][d][u].
// ---------------------------------------------------------------------------
__global__ __launch_bounds__(256)
void splat_kernel(const float* __restrict__ qkv, const float* __restrict__ ln_w,
                  const float* __restrict__ ln_b, const float* __restrict__ centers,
                  const float* __restrict__ outer_radius,
                  const float* __restrict__ hole_radius,
                  const float* __restrict__ entry_points,
                  const float* __restrict__ entry_strengths,
                  const float* __restrict__ amplitude,
                  bf16_t* __restrict__ Fqw, bf16_t* __restrict__ Fk,
                  bf16_t* __restrict__ Vt) {
  const int T = 2048, H = 8, S = 4, E = 4, HD = 64;
  int idx = blockIdx.x * blockDim.x + threadIdx.x;
  if (idx >= 4 * H * T) return;
  int t = idx % T;
  int bh = idx / T;
  int h = bh % H;

  const float* row = qkv + (size_t)((bh / H) * T + t) * 1536;

  // V (bias already added in GEMM), store transposed [d][u] for B-fragments.
  const float* vp = row + 1024 + h * HD;
#pragma unroll 4
  for (int d = 0; d < HD; ++d)
    Vt[((size_t)bh * HD + d) * T + t] = (bf16_t)vp[d];

  for (int which = 0; which < 2; ++which) {
    const float* p = row + which * 512 + h * HD;
    float mu = 0.f;
    for (int d = 0; d < HD; ++d) mu += p[d];
    mu *= (1.f / HD);
    float var = 0.f;
    for (int d = 0; d < HD; ++d) { float df = p[d] - mu; var += df * df; }
    var *= (1.f / HD);
    float inv = rsqrtf(var + 1e-5f);

    float tn[64];
    float tsq = 0.f;
    for (int d = 0; d < HD; ++d) {
      tn[d] = (p[d] - mu) * inv * ln_w[d] + ln_b[d];
      tsq += tn[d] * tn[d];
    }

    bf16_t* out = (which == 0 ? Fqw : Fk) + ((size_t)bh * T + t) * 32;
    for (int s = 0; s < S; ++s) {
      int hs = h * S + s;
      float o_r = fabsf(outer_radius[hs]) + 1e-8f;
      float h_r = fabsf(hole_radius[hs]);
      float ratio = h_r / o_r;
      float wamp = amplitude[hs] * 0.25f;  // amplitude / S

      float tc = 0.f, csq = 0.f;
      const float* cp = centers + (size_t)hs * HD;
      for (int d = 0; d < HD; ++d) { tc += tn[d] * cp[d]; csq += cp[d] * cp[d]; }
      float d2 = fmaxf(tsq - 2.f * tc + csq, 0.f);
      float dd = sqrtf(d2);
      float major = fabsf(dd - o_r);
      float torus = fabsf(sqrtf(major * major + h_r * h_r) - h_r);
      float sd = (ratio < 0.1f) ? major : torus;
      float surf = __expf(-0.5f * sd * sd);
      out[s * 5 + 0] = (bf16_t)(which == 0 ? surf * wamp : surf);

      for (int e = 0; e < E; ++e) {
        const float* ep = entry_points + ((size_t)hs * E + e) * HD;
        float te = 0.f, esq = 0.f;
        for (int d = 0; d < HD; ++d) { te += tn[d] * ep[d]; esq += ep[d] * ep[d]; }
        float ed2 = fmaxf(tsq - 2.f * te + esq, 0.f);
        float ent = __expf(-0.5f * ed2);
        if (which == 0) {
          float sg = 1.f / (1.f + __expf(-entry_strengths[hs * E + e]));
          float wv = (sg > 0.1f ? sg : 0.f) * wamp;
          out[s * 5 + 1 + e] = (bf16_t)(ent * wv);
        } else {
          out[s * 5 + 1 + e] = (bf16_t)ent;
        }
      }
    }
    for (int c = 20; c < 32; ++c) out[c] = (bf16_t)0.f;
  }
}

// ---------------------------------------------------------------------------
// K4: flash attention. grid = (T/128, B*H), block = 256 (8 waves).
// Wave owns 16 query rows; loops over keys in steps of 32.
// scores: 2 WMMA (K=32 padded rank-20); P re-laid out via per-wave LDS;
// P@V: 4 WMMA into fp32 accumulators with online softmax rescale.
// ---------------------------------------------------------------------------
__global__ __launch_bounds__(256)
void attn_kernel(const bf16_t* __restrict__ Fqw, const bf16_t* __restrict__ Fk,
                 const bf16_t* __restrict__ Vt, bf16_t* __restrict__ Obf) {
  const int T = 2048;
  __shared__ bf16_t plds[8 * 16 * 32];

  int lane = threadIdx.x & 31;
  int wid  = threadIdx.x >> 5;
  int hi = lane >> 4, ln = lane & 15;
  int bh = blockIdx.y;
  int t0 = blockIdx.x * 128 + wid * 16;

  const bf16_t* fq_base = Fqw + (size_t)bh * T * 32;
  const bf16_t* fk_base = Fk  + (size_t)bh * T * 32;
  const bf16_t* vt_base = Vt  + (size_t)bh * 64 * T;
  bf16_t* pw = plds + wid * 16 * 32;

  v16bf fq = load_fragA(fq_base, t0, 32, 0, lane);

  v8f o0 = {0,0,0,0,0,0,0,0}, o1 = {0,0,0,0,0,0,0,0};
  v8f o2 = {0,0,0,0,0,0,0,0}, o3 = {0,0,0,0,0,0,0,0};
  float m8[8], l8[8];
#pragma unroll
  for (int r = 0; r < 8; ++r) { m8[r] = -1e30f; l8[r] = 0.f; }

  for (int u0 = 0; u0 < T; u0 += 32) {
    v16bf kb0 = load_fragB(fk_base, u0,      32, 0, lane);
    v16bf kb1 = load_fragB(fk_base, u0 + 16, 32, 0, lane);
    v8f z = {0,0,0,0,0,0,0,0};
    v8f s0 = wmma_bf16(fq, kb0, z);
    v8f s1 = wmma_bf16(fq, kb1, z);

    float sc[8];
#pragma unroll
    for (int r = 0; r < 8; ++r) {
      // row m = r + 8*hi lives on the 16-lane half; reduce across n-lanes.
      float tmax = fmaxf(s0[r], s1[r]);
#pragma unroll
      for (int msk = 8; msk >= 1; msk >>= 1)
        tmax = fmaxf(tmax, __shfl_xor(tmax, msk, 32));
      float mn = fmaxf(m8[r], tmax);
      float scale = __expf(m8[r] - mn);
      float p0 = __expf(s0[r] - mn);
      float p1 = __expf(s1[r] - mn);
      float rs = p0 + p1;
#pragma unroll
      for (int msk = 8; msk >= 1; msk >>= 1)
        rs += __shfl_xor(rs, msk, 32);
      l8[r] = l8[r] * scale + rs;
      m8[r] = mn;
      sc[r] = scale;
      int prow = r + hi * 8;
      pw[prow * 32 + ln]      = (bf16_t)p0;
      pw[prow * 32 + 16 + ln] = (bf16_t)p1;
    }
#pragma unroll
    for (int r = 0; r < 8; ++r) {
      o0[r] *= sc[r]; o1[r] *= sc[r]; o2[r] *= sc[r]; o3[r] *= sc[r];
    }
    asm volatile("s_wait_dscnt 0" ::: "memory");  // LDS P-tile visible in-wave
    v16bf pa = load_fragA(pw, 0, 32, 0, lane);
    o0 = wmma_bf16(pa, load_fragB(vt_base, 0,  T, u0, lane), o0);
    o1 = wmma_bf16(pa, load_fragB(vt_base, 16, T, u0, lane), o1);
    o2 = wmma_bf16(pa, load_fragB(vt_base, 32, T, u0, lane), o2);
    o3 = wmma_bf16(pa, load_fragB(vt_base, 48, T, u0, lane), o3);
  }

  int b = bh >> 3, h = bh & 7;
#pragma unroll
  for (int r = 0; r < 8; ++r) {
    float inv = 1.f / l8[r];
    size_t mrow = (size_t)(b * T + t0 + r + hi * 8) * 512 + h * 64;
    Obf[mrow + 0  + ln] = (bf16_t)(o0[r] * inv);
    Obf[mrow + 16 + ln] = (bf16_t)(o1[r] * inv);
    Obf[mrow + 32 + ln] = (bf16_t)(o2[r] * inv);
    Obf[mrow + 48 + ln] = (bf16_t)(o3[r] * inv);
  }
}

// ---------------------------------------------------------------------------
extern "C" void kernel_launch(void* const* d_in, const int* in_sizes, int n_in,
                              void* d_out, int out_size, void* d_ws, size_t ws_size,
                              hipStream_t stream) {
  const int B = 4, T = 2048, D = 512, H = 8;
  const int M = B * T;          // 8192
  const int N_QKV = 3 * D;      // 1536

  const float* x        = (const float*)d_in[0];
  const float* qkv_w    = (const float*)d_in[1];
  const float* qkv_b    = (const float*)d_in[2];
  const float* ln_w     = (const float*)d_in[3];
  const float* ln_b     = (const float*)d_in[4];
  const float* centers  = (const float*)d_in[5];
  const float* outer_r  = (const float*)d_in[6];
  const float* hole_r   = (const float*)d_in[7];
  const float* entry_p  = (const float*)d_in[8];
  const float* entry_s  = (const float*)d_in[9];
  const float* amp      = (const float*)d_in[10];
  const float* out_w    = (const float*)d_in[11];
  const float* out_b    = (const float*)d_in[12];
  float* out            = (float*)d_out;

  // Workspace carve-up (all chunks 256B-aligned by construction).
  char* ws = (char*)d_ws;
  bf16_t* x_bf  = (bf16_t*)ws;                  ws += (size_t)M * D * 2;        // 8 MiB
  bf16_t* wq_bf = (bf16_t*)ws;                  ws += (size_t)N_QKV * D * 2;    // 1.5 MiB
  bf16_t* ow_bf = (bf16_t*)ws;                  ws += (size_t)D * D * 2;        // 0.5 MiB
  float*  qkv   = (float*)ws;                   ws += (size_t)M * N_QKV * 4;    // 48 MiB
  bf16_t* Fqw   = (bf16_t*)ws;                  ws += (size_t)B * H * T * 32 * 2;
  bf16_t* Fk    = (bf16_t*)ws;                  ws += (size_t)B * H * T * 32 * 2;
  bf16_t* Vt    = (bf16_t*)ws;                  ws += (size_t)B * H * 64 * T * 2;
  bf16_t* Abf   = (bf16_t*)ws;                  ws += (size_t)M * D * 2;        // 8 MiB

  // K1: converts
  {
    int n0 = M * D;
    cvt_bf16_kernel<<<(n0 + 255) / 256, 256, 0, stream>>>(x, x_bf, n0);
    int n1 = N_QKV * D;
    cvt_bf16_kernel<<<(n1 + 255) / 256, 256, 0, stream>>>(qkv_w, wq_bf, n1);
    int n2 = D * D;
    cvt_bf16_kernel<<<(n2 + 255) / 256, 256, 0, stream>>>(out_w, ow_bf, n2);
  }

  // K2: qkv = x @ qkv_w^T + qkv_b  (8192 x 1536 x 512)
  {
    dim3 grid(N_QKV / 128, M / 64);
    gemm_bf16_kernel<<<grid, 256, 0, stream>>>(x_bf, wq_bf, qkv_b, qkv,
                                               M, N_QKV, D);
  }

  // K3: LN + splat factors + V transpose
  {
    int total = B * H * T;
    splat_kernel<<<total / 256, 256, 0, stream>>>(qkv, ln_w, ln_b, centers,
                                                  outer_r, hole_r, entry_p,
                                                  entry_s, amp, Fqw, Fk, Vt);
  }

  // K4: flash attention -> Abf (B,T,D) bf16
  {
    dim3 grid(T / 128, B * H);
    attn_kernel<<<grid, 256, 0, stream>>>(Fqw, Fk, Vt, Abf);
  }

  // K5: out = Abf @ out_w^T + out_b  (8192 x 512 x 512)
  {
    dim3 grid(D / 128, M / 64);
    gemm_bf16_kernel<<<grid, 256, 0, stream>>>(Abf, ow_bf, out_b, out,
                                               M, D, D);
  }

  (void)in_sizes; (void)n_in; (void)out_size; (void)ws_size;
}
